// PointPillarScatter_Agg_Memory_1_scale_64553358459106
// MI455X (gfx1250) — matrix operand order repaired and checked by reading, not compiled, not run on hardware
//
#include <hip/hip_runtime.h>
#include <hip/hip_bf16.h>

// ---------------------------------------------------------------------------
// Problem constants (from reference)
// ---------------------------------------------------------------------------
constexpr int Bn  = 2;
constexpr int Pn  = 4000;
constexpr int NPn = 4096;
constexpr int Dn  = 64;
constexpr int DSn = 64;
constexpr int Mn  = 512;
constexpr int NXn = 432;
constexpr int NYn = 496;
constexpr int HWn = NXn * NYn;      // 214272

// output layout (floats), concatenated in reference return order
constexpr size_t SZ_SP   = (size_t)Bn * 128 * HWn;    // spatial_features
constexpr size_t OFF_SP  = 0;
constexpr size_t OFF_SPP = SZ_SP;                      // spatial_features_point
constexpr size_t OFF_SC  = 2 * SZ_SP;                  // spatial_scale_features
constexpr size_t OFF_PP  = OFF_SC + (size_t)Bn * DSn * HWn;  // point_positive
constexpr size_t OFF_PM  = OFF_PP + (size_t)Bn * Pn * Dn;    // memory_positive

constexpr int WAVES = 4;            // waves per block
constexpr int TPB   = WAVES * 32;   // 128 threads
constexpr int PPB   = WAVES * 16;   // 64 pillars per block

typedef unsigned short u16;
typedef __attribute__((ext_vector_type(16))) __bf16 v16bf;
typedef __attribute__((ext_vector_type(8)))  float  v8f;

struct QPair { uint4 lo; uint4 hi; };   // 32 bytes == v16bf

__device__ inline v16bf load_frag16(const u16* p1, const u16* p2) {
    QPair q;
    q.lo = *reinterpret_cast<const uint4*>(p1);
    q.hi = *reinterpret_cast<const uint4*>(p2);
    return __builtin_bit_cast(v16bf, q);
}

__device__ inline u16 f2bf(float f) {
    unsigned u = __builtin_bit_cast(unsigned, f);
    u += 0x7FFFu + ((u >> 16) & 1u);       // round-to-nearest-even
    return (u16)(u >> 16);
}

// ---------------------------------------------------------------------------
// fp32 -> bf16 conversion of pillars, points, W into workspace
// ---------------------------------------------------------------------------
__global__ void convert_bf16_kernel(const float* __restrict__ pil,
                                    const float* __restrict__ pts,
                                    const float* __restrict__ W,
                                    u16* __restrict__ pilB,
                                    u16* __restrict__ ptsB,
                                    u16* __restrict__ WB) {
    const int n1 = Bn * Pn * Dn;
    const int n2 = Bn * NPn * Dn;
    const int n3 = Mn * Dn;
    int t = blockIdx.x * blockDim.x + threadIdx.x;
    if (t < n1)                { pilB[t] = f2bf(pil[t]); return; }
    t -= n1;
    if (t < n2)                { ptsB[t] = f2bf(pts[t]); return; }
    t -= n2;
    if (t < n3)                { WB[t]  = f2bf(W[t]); }
}

// ---------------------------------------------------------------------------
// Deterministic duplicate-index resolution: largest pillar id wins a column
// ---------------------------------------------------------------------------
__global__ void winner_kernel(const int* __restrict__ indices,
                              int* __restrict__ winner) {
    int t = blockIdx.x * blockDim.x + threadIdx.x;
    if (t >= Bn * Pn) return;
    int b = t / Pn, p = t % Pn;
    atomicMax(&winner[(size_t)b * HWn + indices[t]], p);
}

// ---------------------------------------------------------------------------
// Top-8 maintenance (sorted ascending, static indexing only)
// ---------------------------------------------------------------------------
__device__ inline void top8_insert(float (&tv)[8], int (&ti)[8], float val, int g) {
    if (val > tv[0]) {
        tv[0] = val; ti[0] = g;
#pragma unroll
        for (int s = 0; s < 7; ++s) {
            if (tv[s] > tv[s + 1]) {
                float tf = tv[s]; tv[s] = tv[s + 1]; tv[s + 1] = tf;
                int   tg = ti[s]; ti[s] = ti[s + 1]; ti[s + 1] = tg;
            }
        }
    }
}

// ---------------------------------------------------------------------------
// Attention kernel: 16 pillars per wave. WMMA-GEMM pillars x srcT with
// ping-pong double-buffered B-fragment loads (2x unrolled, no rotation
// copies), running top-8 logits per pillar (chunk-max fast-reject),
// softmax over the 8, weighted-sum of src rows -> pos vector; scatter
// [pillars ; pos] into grid + write pos rows.
// ---------------------------------------------------------------------------
template <int NCOLS>
__global__ __launch_bounds__(TPB) void attn_kernel(
        const float* __restrict__ pillarsF,   // [B,P,64]
        const float* __restrict__ srcF,       // [B?,NCOLS,64] (W has no B dim)
        const u16*   __restrict__ pilB,       // [B,P,64] bf16
        const u16*   __restrict__ srcB,       // bf16
        const int*   __restrict__ indices,    // [B,P]
        const int*   __restrict__ winner,     // [B,HW]
        float*       __restrict__ out,
        size_t off_grid,                      // OFF_SPP or OFF_SP
        size_t off_pos,                       // OFF_PP or OFF_PM
        int src_has_batch)
{
    constexpr int NCH = NCOLS / 16;           // 256 or 32 (even)

    __shared__ float tile[WAVES][16][16];
    __shared__ float mval[WAVES][16][8];
    __shared__ int   midx[WAVES][16][8];

    const int b    = blockIdx.y;
    const int wave = threadIdx.x >> 5;
    const int lane = threadIdx.x & 31;
    const int half = lane >> 4;
    const int m    = lane & 15;

    const int rowP = blockIdx.x * PPB + wave * 16 + m;
    const int rowC = rowP < Pn ? rowP : Pn - 1;

    const u16* pil = pilB + (size_t)b * Pn * Dn;
    const u16* src = srcB + (src_has_batch ? (size_t)b * NCOLS * Dn : 0);

    // A fragments (constant over the loop): 16x32 bf16, row m per lane-pair.
    // lane(half,m), element j -> K = (j%8) + 8*half + 16*(j/8) + kb
    const u16* arow = pil + (size_t)rowC * Dn;
    const v16bf a0 = load_frag16(arow +  8 * half,      arow + 16 +  8 * half);
    const v16bf a1 = load_frag16(arow + 32 + 8 * half,  arow + 48 +  8 * half);

    float tv[8]; int ti[8];
#pragma unroll
    for (int k = 0; k < 8; ++k) { tv[k] = -3.0e38f; ti[k] = 0; }

    // B fragment: B[K,N] = src[col0+N][K]; lane(half,n=m), elem j -> K=(j%8)+8*half+16*(j/8)+kb
    auto loadB = [&](int ch, v16bf& x, v16bf& y) {
        const u16* nb = src + (size_t)(ch * 16 + m) * Dn + 8 * half;
        x = load_frag16(nb,      nb + 16);
        y = load_frag16(nb + 32, nb + 48);
    };

    // per-chunk: WMMA tile -> LDS transpose -> top-8 scan with fast reject
    auto scan = [&](v8f c, int col0) {
#pragma unroll
        for (int v = 0; v < 8; ++v)
            tile[wave][v + 8 * half][m] = c[v];

        asm volatile("s_wait_dscnt 0x0" ::: "memory");

        float vals[8];
#pragma unroll
        for (int j = 0; j < 8; ++j)
            vals[j] = tile[wave][m][2 * j + half];

        float cmax = fmaxf(fmaxf(fmaxf(vals[0], vals[1]), fmaxf(vals[2], vals[3])),
                           fmaxf(fmaxf(vals[4], vals[5]), fmaxf(vals[6], vals[7])));
        if (cmax > tv[0]) {
#pragma unroll
            for (int j = 0; j < 8; ++j)
                top8_insert(tv, ti, vals[j], col0 + 2 * j + half);
        }
    };

    v16bf b0a, b1a, b0b, b1b;                 // ping-pong buffers
    loadB(0, b0a, b1a);
    loadB(1, b0b, b1b);

    for (int ch = 0; ch < NCH; ch += 2) {
        {   // even chunk: consume A-buffer, refill with ch+2
            v8f c = {};
            c = __builtin_amdgcn_wmma_f32_16x16x32_bf16(false, a0, false, b0a,
                                                        (short)0, c, false, false);
            c = __builtin_amdgcn_wmma_f32_16x16x32_bf16(false, a1, false, b1a,
                                                        (short)0, c, false, false);
            loadB(ch + 2 < NCH ? ch + 2 : 0, b0a, b1a);
            scan(c, ch * 16);
        }
        {   // odd chunk: consume B-buffer, refill with ch+3
            v8f c = {};
            c = __builtin_amdgcn_wmma_f32_16x16x32_bf16(false, a0, false, b0b,
                                                        (short)0, c, false, false);
            c = __builtin_amdgcn_wmma_f32_16x16x32_bf16(false, a1, false, b1b,
                                                        (short)0, c, false, false);
            loadB(ch + 3 < NCH ? ch + 3 : 1, b0b, b1b);
            scan(c, ch * 16 + 16);
        }
    }

    // merge the two per-row candidate lists (lane m and lane m+16)
    if (half == 1) {
#pragma unroll
        for (int k = 0; k < 8; ++k) { mval[wave][m][k] = tv[k]; midx[wave][m][k] = ti[k]; }
    }
    asm volatile("s_wait_dscnt 0x0" ::: "memory");

    if (half == 0) {
#pragma unroll
        for (int k = 0; k < 8; ++k)
            top8_insert(tv, ti, mval[wave][m][k], midx[wave][m][k]);

        // softmax over the 8 retained logits (tv sorted ascending; max = tv[7])
        float w[8];
        float mx = tv[7], s = 0.f;
#pragma unroll
        for (int k = 0; k < 8; ++k) { w[k] = __expf(tv[k] - mx); s += w[k]; }
        float inv = 1.f / s;
#pragma unroll
        for (int k = 0; k < 8; ++k) w[k] *= inv;

        if (rowP < Pn) {
            const float* sf  = srcF + (src_has_batch ? (size_t)b * NCOLS * Dn : 0);
            const float* plf = pillarsF + (size_t)b * Pn * Dn + (size_t)rowP * Dn;
            const int col = indices[b * Pn + rowP];
            const bool win = (winner[(size_t)b * HWn + col] == rowP);
            float* opos = out + off_pos + (size_t)(b * Pn + rowP) * Dn;
            float* grid = out + off_grid + (size_t)b * 128 * HWn;

            for (int dc = 0; dc < 16; ++dc) {          // 16 x float4 = 64 dims
                float4 acc = {0.f, 0.f, 0.f, 0.f};
#pragma unroll
                for (int k = 0; k < 8; ++k) {
                    float4 pv = *reinterpret_cast<const float4*>(
                        sf + (size_t)ti[k] * Dn + dc * 4);
                    acc.x += w[k] * pv.x; acc.y += w[k] * pv.y;
                    acc.z += w[k] * pv.z; acc.w += w[k] * pv.w;
                }
                *reinterpret_cast<float4*>(opos + dc * 4) = acc;
                if (win) {
                    grid[(size_t)(64 + dc * 4 + 0) * HWn + col] = acc.x;
                    grid[(size_t)(64 + dc * 4 + 1) * HWn + col] = acc.y;
                    grid[(size_t)(64 + dc * 4 + 2) * HWn + col] = acc.z;
                    grid[(size_t)(64 + dc * 4 + 3) * HWn + col] = acc.w;
                }
            }
            if (win) {
                for (int d = 0; d < 64; ++d)
                    grid[(size_t)d * HWn + col] = plf[d];
            }
        }
    }
}

// ---------------------------------------------------------------------------
// scale scatter: sp_scale[:, idx[p]] = scale[p,:]
// ---------------------------------------------------------------------------
__global__ void scale_scatter_kernel(const float* __restrict__ scale,
                                     const int* __restrict__ indices,
                                     const int* __restrict__ winner,
                                     float* __restrict__ out) {
    int t = blockIdx.x * blockDim.x + threadIdx.x;
    if (t >= Bn * Pn) return;
    int b = t / Pn, p = t % Pn;
    int col = indices[t];
    if (winner[(size_t)b * HWn + col] != p) return;
    const float* s = scale + (size_t)t * DSn;
    float* o = out + OFF_SC + (size_t)b * DSn * HWn + col;
    for (int d = 0; d < DSn; ++d)
        o[(size_t)d * HWn] = s[d];
}

// ---------------------------------------------------------------------------
extern "C" void kernel_launch(void* const* d_in, const int* in_sizes, int n_in,
                              void* d_out, int out_size, void* d_ws, size_t ws_size,
                              hipStream_t stream) {
    const float* pillars = (const float*)d_in[0];   // [B,P,64]
    const float* scale   = (const float*)d_in[1];   // [B,P,64]
    const float* points  = (const float*)d_in[2];   // [B,NP,64]
    const float* W       = (const float*)d_in[3];   // [M,64]
    const int*   indices = (const int*)d_in[4];     // [B,P]
    float* out = (float*)d_out;

    // workspace carve-out
    char* ws = (char*)d_ws;
    size_t o = 0;
    u16* pilB = (u16*)(ws + o); o += (size_t)Bn * Pn  * Dn * sizeof(u16);
    u16* ptsB = (u16*)(ws + o); o += (size_t)Bn * NPn * Dn * sizeof(u16);
    u16* WB   = (u16*)(ws + o); o += (size_t)Mn * Dn * sizeof(u16);
    int* winner = (int*)(ws + o);

    // zero the three dense grids (pos_point / pos_mem are fully overwritten)
    hipMemsetAsync(out, 0, OFF_PP * sizeof(float), stream);
    hipMemsetAsync(winner, 0xFF, (size_t)Bn * HWn * sizeof(int), stream);

    {
        int total = Bn * Pn * Dn + Bn * NPn * Dn + Mn * Dn;
        convert_bf16_kernel<<<(total + 255) / 256, 256, 0, stream>>>(
            pillars, points, W, pilB, ptsB, WB);
    }
    winner_kernel<<<(Bn * Pn + 255) / 256, 256, 0, stream>>>(indices, winner);

    dim3 grid((Pn + PPB - 1) / PPB, Bn);   // 63 x 2 blocks, 4 waves x 16 pillars
    attn_kernel<NPn><<<grid, TPB, 0, stream>>>(
        pillars, points, pilB, ptsB, indices, winner, out,
        OFF_SPP, OFF_PP, /*src_has_batch=*/1);
    attn_kernel<Mn><<<grid, TPB, 0, stream>>>(
        pillars, W, pilB, WB, indices, winner, out,
        OFF_SP, OFF_PM, /*src_has_batch=*/0);

    scale_scatter_kernel<<<(Bn * Pn + 255) / 256, 256, 0, stream>>>(
        scale, indices, winner, out);
}
